// NodeModel_61684320305462
// MI455X (gfx1250) — matrix operand order repaired and checked by reading, not compiled, not run on hardware
//
#include <hip/hip_runtime.h>

#define NN 100000
#define EE 1600000
#define FF 64
#define BN_EPS 1e-5f

typedef __attribute__((ext_vector_type(16))) __bf16 v16bf;
typedef __attribute__((ext_vector_type(8)))  __bf16 v8bf;
typedef __attribute__((ext_vector_type(8)))  float  v8f;
typedef __attribute__((ext_vector_type(4)))  float  v4f;

static __device__ __forceinline__ v8f wmma_bf16(v16bf a, v16bf b, v8f c) {
  return __builtin_amdgcn_wmma_f32_16x16x32_bf16(false, a, false, b, (short)0, c, false, false);
}

static __device__ __forceinline__ v16bf frag_join(const __bf16* lo, const __bf16* hi) {
  v8bf a = *(const v8bf*)lo;
  v8bf b = *(const v8bf*)hi;
  v16bf r;
#pragma unroll
  for (int i = 0; i < 8; ++i) { r[i] = a[i]; r[i + 8] = b[i]; }
  return r;
}

// A fragment: 16x32 bf16 from row-major [M][stride] buffer.
// lane<16: elems 0..7 = K k0..k0+7, elems 8..15 = K k0+16..k0+23 (row = m0+lane)
// lane>=16: same +8 on K, row = m0+lane-16
static __device__ __forceinline__ v16bf load_a_frag(const __bf16* base, int stride,
                                                    int m0, int k0, int lane) {
  const __bf16* p = base + (m0 + (lane & 15)) * stride + k0 + ((lane >> 4) << 3);
  return frag_join(p, p + 16);
}

// B fragment: 32x16 bf16 from transposed weights wT[N][K] (row stride K elems).
// lane<16: col n0+lane, elems 0..7 = K k0..k0+7, elems 8..15 = K k0+16..k0+23; lane>=16: same +8 on K
static __device__ __forceinline__ v16bf load_b_frag(const __bf16* wT, int K,
                                                    int n0, int k0, int lane) {
  const __bf16* p = wT + (n0 + (lane & 15)) * K + k0 + ((lane >> 4) << 3);
  return frag_join(p, p + 16);
}

// ---------------- utility kernels ----------------

__global__ void zero_f32_kernel(float* __restrict__ p, int n) {
  for (int i = blockIdx.x * blockDim.x + threadIdx.x; i < n; i += gridDim.x * blockDim.x)
    p[i] = 0.0f;
}

// src row-major [K][N] f32 -> dst row-major [N][K] bf16 (transposed, coalesced writes)
__global__ void transpose_bf16_kernel(const float* __restrict__ src, __bf16* __restrict__ dst,
                                      int K, int N) {
  int i = blockIdx.x * blockDim.x + threadIdx.x;
  if (i >= K * N) return;
  int n = i / K, k = i - n * K;
  dst[i] = (__bf16)src[k * N + n];
}

// ---------------- edge (message) kernel ----------------

__global__ __launch_bounds__(256) void edge_msg_kernel(
    const float* __restrict__ x, const int* __restrict__ ei,
    const float* __restrict__ ea, const int* __restrict__ is_src_p,
    const __bf16* __restrict__ wT1, const float* __restrict__ b1,
    const __bf16* __restrict__ wT2, const float* __restrict__ b2,
    float* __restrict__ agg, float* __restrict__ cnt) {
  __shared__ __bf16 sA[64 * 136];
  __shared__ __bf16 sH[64 * 136];
  __shared__ int sScat[64];

  const int tid = threadIdx.x, lane = tid & 31, wv = tid >> 5;
  const int is_src = *is_src_p;
  const int* gidx = ei + (is_src ? EE : 0);  // gather = tgt if is_source
  const int* sidx = ei + (is_src ? 0 : EE);  // scatter = src if is_source
  const int e0 = blockIdx.x * 64;

  // ---- stage A = [x[gather] | edge_attr] as bf16 (4 threads / row, 32 cols each)
  {
    const int row = tid >> 2, t4 = tid & 3;
    const int e = e0 + row;
    __bf16* dst = sA + row * 136 + t4 * 32;
    if (e < EE) {
      const float* srcp;
      if (t4 < 2) { const int g = gidx[e]; srcp = x + g * FF + t4 * 32; }
      else        { srcp = ea + e * FF + (t4 - 2) * 32; }
#pragma unroll
      for (int j = 0; j < 32; j += 4) {
        v4f v = *(const v4f*)(srcp + j);
        dst[j + 0] = (__bf16)v[0]; dst[j + 1] = (__bf16)v[1];
        dst[j + 2] = (__bf16)v[2]; dst[j + 3] = (__bf16)v[3];
      }
    } else {
#pragma unroll
      for (int j = 0; j < 32; ++j) dst[j] = (__bf16)0.0f;
    }
    if (tid < 64) {
      const int e2 = e0 + tid;
      const int s = (e2 < EE) ? sidx[e2] : -1;
      sScat[tid] = s;
      if (s >= 0) atomicAdd(&cnt[s], 1.0f);  // degree for mean aggregation
    }
  }
  __syncthreads();

  // ---- layer 1: [64,128] @ w1[128,128] -> ReLU -> bf16 hidden in LDS
  {
    const int n0 = wv * 16;
    const v16bf B0 = load_b_frag(wT1, 128, n0, 0, lane);
    const v16bf B1 = load_b_frag(wT1, 128, n0, 32, lane);
    const v16bf B2 = load_b_frag(wT1, 128, n0, 64, lane);
    const v16bf B3 = load_b_frag(wT1, 128, n0, 96, lane);
    const float bias = b1[n0 + (lane & 15)];
    const int col = n0 + (lane & 15);
#pragma unroll
    for (int mi = 0; mi < 4; ++mi) {
      v8f acc = {};
      acc = wmma_bf16(load_a_frag(sA, 136, mi * 16, 0, lane), B0, acc);
      acc = wmma_bf16(load_a_frag(sA, 136, mi * 16, 32, lane), B1, acc);
      acc = wmma_bf16(load_a_frag(sA, 136, mi * 16, 64, lane), B2, acc);
      acc = wmma_bf16(load_a_frag(sA, 136, mi * 16, 96, lane), B3, acc);
      const int rbase = mi * 16 + ((lane >> 4) << 3);
#pragma unroll
      for (int r = 0; r < 8; ++r) {
        float v = acc[r] + bias;
        v = v > 0.0f ? v : 0.0f;
        sH[(rbase + r) * 136 + col] = (__bf16)v;
      }
    }
  }
  __syncthreads();

  // ---- layer 2: [64,128] @ w2[128,64] + b2, scatter-atomic into agg
  {
#pragma unroll
    for (int s = 0; s < 2; ++s) {
      const int st = wv * 2 + s;
      const int m0 = (st >> 2) * 16, n0 = (st & 3) * 16;
      v8f acc = {};
      acc = wmma_bf16(load_a_frag(sH, 136, m0, 0, lane),
                      load_b_frag(wT2, 128, n0, 0, lane), acc);
      acc = wmma_bf16(load_a_frag(sH, 136, m0, 32, lane),
                      load_b_frag(wT2, 128, n0, 32, lane), acc);
      acc = wmma_bf16(load_a_frag(sH, 136, m0, 64, lane),
                      load_b_frag(wT2, 128, n0, 64, lane), acc);
      acc = wmma_bf16(load_a_frag(sH, 136, m0, 96, lane),
                      load_b_frag(wT2, 128, n0, 96, lane), acc);
      const float bias = b2[n0 + (lane & 15)];
      const int n = n0 + (lane & 15);
      const int rbase = m0 + ((lane >> 4) << 3);
#pragma unroll
      for (int r = 0; r < 8; ++r) {
        const int d = sScat[rbase + r];
        if (d >= 0) atomicAdd(&agg[d * FF + n], acc[r] + bias);
      }
    }
  }
}

// ---------------- node (update) kernel ----------------

__global__ __launch_bounds__(256) void node_update_kernel(
    const float* __restrict__ x, const float* __restrict__ agg,
    const float* __restrict__ cnt, const int* __restrict__ is_src_p,
    const __bf16* __restrict__ uT1, const float* __restrict__ ub1,
    const __bf16* __restrict__ uT2, const float* __restrict__ ub2,
    float* __restrict__ hout, float* __restrict__ gsum, float* __restrict__ gsumsq) {
  __shared__ __bf16 sA[64 * 136];
  __shared__ __bf16 sH[64 * 136];

  const int tid = threadIdx.x, lane = tid & 31, wv = tid >> 5;
  const int is_src = *is_src_p;
  const int nb = blockIdx.x * 64;

  // ---- stage A = [x | agg(/cnt if mean)] as bf16
  {
    const int row = tid >> 2, t4 = tid & 3;
    const int node = nb + row;
    __bf16* dst = sA + row * 136 + t4 * 32;
    if (node < NN) {
      if (t4 < 2) {
        const float* sp = x + node * FF + t4 * 32;
#pragma unroll
        for (int j = 0; j < 32; j += 4) {
          v4f v = *(const v4f*)(sp + j);
          dst[j + 0] = (__bf16)v[0]; dst[j + 1] = (__bf16)v[1];
          dst[j + 2] = (__bf16)v[2]; dst[j + 3] = (__bf16)v[3];
        }
      } else {
        const float c = cnt[node];
        const float scale = is_src ? (1.0f / fmaxf(c, 1.0f)) : 1.0f;
        const float* sp = agg + node * FF + (t4 - 2) * 32;
#pragma unroll
        for (int j = 0; j < 32; j += 4) {
          v4f v = *(const v4f*)(sp + j);
          dst[j + 0] = (__bf16)(v[0] * scale); dst[j + 1] = (__bf16)(v[1] * scale);
          dst[j + 2] = (__bf16)(v[2] * scale); dst[j + 3] = (__bf16)(v[3] * scale);
        }
      }
    } else {
#pragma unroll
      for (int j = 0; j < 32; ++j) dst[j] = (__bf16)0.0f;
    }
  }
  __syncthreads();

  // ---- layer 1: [64,128] @ u1[128,64] -> ReLU -> bf16 hidden
  {
#pragma unroll
    for (int s = 0; s < 2; ++s) {
      const int st = wv * 2 + s;
      const int m0 = (st >> 2) * 16, n0 = (st & 3) * 16;
      v8f acc = {};
      acc = wmma_bf16(load_a_frag(sA, 136, m0, 0, lane),
                      load_b_frag(uT1, 128, n0, 0, lane), acc);
      acc = wmma_bf16(load_a_frag(sA, 136, m0, 32, lane),
                      load_b_frag(uT1, 128, n0, 32, lane), acc);
      acc = wmma_bf16(load_a_frag(sA, 136, m0, 64, lane),
                      load_b_frag(uT1, 128, n0, 64, lane), acc);
      acc = wmma_bf16(load_a_frag(sA, 136, m0, 96, lane),
                      load_b_frag(uT1, 128, n0, 96, lane), acc);
      const float bias = ub1[n0 + (lane & 15)];
      const int col = n0 + (lane & 15);
      const int rbase = m0 + ((lane >> 4) << 3);
#pragma unroll
      for (int r = 0; r < 8; ++r) {
        float v = acc[r] + bias;
        v = v > 0.0f ? v : 0.0f;
        sH[(rbase + r) * 136 + col] = (__bf16)v;
      }
    }
  }
  __syncthreads();

  // ---- layer 2: [64,64] @ u2[64,64] + ub2; write pre-BN h, accumulate BN stats
  {
#pragma unroll
    for (int s = 0; s < 2; ++s) {
      const int st = wv * 2 + s;
      const int m0 = (st >> 2) * 16, n0 = (st & 3) * 16;
      v8f acc = {};
      acc = wmma_bf16(load_a_frag(sH, 136, m0, 0, lane),
                      load_b_frag(uT2, 64, n0, 0, lane), acc);
      acc = wmma_bf16(load_a_frag(sH, 136, m0, 32, lane),
                      load_b_frag(uT2, 64, n0, 32, lane), acc);
      const float bias = ub2[n0 + (lane & 15)];
      const int n = n0 + (lane & 15);
      const int rbase = m0 + ((lane >> 4) << 3);
      float ps = 0.0f, ps2 = 0.0f;
#pragma unroll
      for (int r = 0; r < 8; ++r) {
        const int node = nb + rbase + r;
        if (node < NN) {
          const float h = acc[r] + bias;
          hout[node * FF + n] = h;
          ps += h; ps2 += h * h;
        }
      }
      atomicAdd(&gsum[n], ps);
      atomicAdd(&gsumsq[n], ps2);
    }
  }
}

// ---------------- BatchNorm finalize ----------------

__global__ void bn_finalize_kernel(float* __restrict__ out,
                                   const float* __restrict__ gsum,
                                   const float* __restrict__ gsumsq,
                                   const float* __restrict__ gamma,
                                   const float* __restrict__ beta) {
  const int total = NN * FF;
  for (int i = blockIdx.x * blockDim.x + threadIdx.x; i < total; i += gridDim.x * blockDim.x) {
    const int c = i & (FF - 1);
    const float mu = gsum[c] * (1.0f / NN);
    const float var = gsumsq[c] * (1.0f / NN) - mu * mu;  // biased, as torch BN
    out[i] = gamma[c] * (out[i] - mu) * rsqrtf(var + BN_EPS) + beta[c];
  }
}

// ---------------- launch ----------------

extern "C" void kernel_launch(void* const* d_in, const int* in_sizes, int n_in,
                              void* d_out, int out_size, void* d_ws, size_t ws_size,
                              hipStream_t stream) {
  const float* x     = (const float*)d_in[0];
  const int*   ei    = (const int*)d_in[1];
  const float* ea    = (const float*)d_in[2];
  const int*   issrc = (const int*)d_in[3];
  const float* w1    = (const float*)d_in[4];
  const float* b1    = (const float*)d_in[5];
  const float* w2    = (const float*)d_in[6];
  const float* b2    = (const float*)d_in[7];
  const float* u1    = (const float*)d_in[8];
  const float* ub1   = (const float*)d_in[9];
  const float* u2    = (const float*)d_in[10];
  const float* ub2   = (const float*)d_in[11];
  const float* gamma = (const float*)d_in[12];
  const float* beta  = (const float*)d_in[13];

  // workspace layout (floats, then bf16 weights)
  float* wsf    = (float*)d_ws;
  float* agg    = wsf;                       // NN*FF
  float* cnt    = wsf + 6400000;             // NN
  float* gsum   = wsf + 6500000;             // FF
  float* gsumsq = wsf + 6500064;             // FF
  __bf16* wT1 = (__bf16*)((char*)d_ws + 26000512);  // [128][128]
  __bf16* wT2 = wT1 + 128 * 128;                    // [64][128]
  __bf16* uT1 = wT2 + 64 * 128;                     // [64][128]
  __bf16* uT2 = uT1 + 64 * 128;                     // [64][64]

  zero_f32_kernel<<<2048, 256, 0, stream>>>(wsf, 6500128);

  transpose_bf16_kernel<<<64, 256, 0, stream>>>(w1, wT1, 128, 128);
  transpose_bf16_kernel<<<32, 256, 0, stream>>>(w2, wT2, 128, 64);
  transpose_bf16_kernel<<<32, 256, 0, stream>>>(u1, uT1, 128, 64);
  transpose_bf16_kernel<<<16, 256, 0, stream>>>(u2, uT2, 64, 64);

  edge_msg_kernel<<<(EE + 63) / 64, 256, 0, stream>>>(x, ei, ea, issrc, wT1, b1, wT2, b2, agg, cnt);

  node_update_kernel<<<(NN + 63) / 64, 256, 0, stream>>>(x, agg, cnt, issrc,
                                                         uT1, ub1, uT2, ub2,
                                                         (float*)d_out, gsum, gsumsq);

  bn_finalize_kernel<<<1024, 256, 0, stream>>>((float*)d_out, gsum, gsumsq, gamma, beta);
}
